// Ranking_23287312679666
// MI455X (gfx1250) — matrix operand order repaired and checked by reading, not compiled, not run on hardware
//
#include <hip/hip_runtime.h>
#include <hip/hip_bf16.h>
#include <math.h>

#define HD 16
#define EPSBN 1e-5f
#define WPB 8             // waves per block in the GRU kernel
#define GRUBLK (WPB * 32) // 256 threads = 8 wave32

typedef _Float16 v16h __attribute__((ext_vector_type(16)));
typedef _Float16 v8h  __attribute__((ext_vector_type(8)));
typedef float    v8f  __attribute__((ext_vector_type(8)));

// Fast activations: v_exp_f32 + v_rcp_f32 only (no IEEE divide expansion).
__device__ __forceinline__ float sigmf(float x) {
    return __builtin_amdgcn_rcpf(1.0f + __expf(-x));
}
__device__ __forceinline__ float tanhfast(float x) {
    float e = __expf(2.0f * x);
    return 1.0f - 2.0f * __builtin_amdgcn_rcpf(e + 1.0f);
}

// -----------------------------------------------------------------------------
// Kernel 1: embedding gather + GRU (WMMA) + degenerate-attention epilogue.
// One wave handles a 16-batch tile. Per step: A = [x | h] (16x32 f16),
// 4x v_wmma_f32_16x16x32_f16 against weight fragments held in VGPRs.
// Writes all_enc[B,48] and atomically accumulates BN1 column sum/sumsq.
// -----------------------------------------------------------------------------
__global__ void __launch_bounds__(GRUBLK)
k_gru_attn(const int* __restrict__ user_id,
           const int* __restrict__ hist_item,
           const int* __restrict__ hist_cat,
           const int* __restrict__ tgt_item,
           const int* __restrict__ tgt_cat,
           const float* __restrict__ user_table,
           const float* __restrict__ item_table,
           const float* __restrict__ cat_table,
           const float* __restrict__ Wih, const float* __restrict__ Whh,
           const float* __restrict__ bih, const float* __restrict__ bhh,
           const float* __restrict__ Win, const float* __restrict__ binp,
           const float* __restrict__ Wout, const float* __restrict__ bout,
           float* __restrict__ all_enc, float* __restrict__ g_stats1,
           int B, int T)
{
    __shared__ __align__(16) _Float16 h_lds[WPB][16][16]; // per-wave h mirror (f16)
    __shared__ float tgt_lds[WPB][16][16];                // per-wave target emb tile
    __shared__ float stat_lds[96];                        // 48 sums + 48 sumsq

    const int  tid  = threadIdx.x;
    const int  wave = tid >> 5;
    const int  lane = tid & 31;
    const int  n    = lane & 15;        // column (hidden dim / gate) owned in C/D layout
    const bool lo   = (lane < 16);
    const int  b0   = (blockIdx.x * WPB + wave) * 16;
    const int  r    = n;                // A-matrix row (batch row) this lane gathers
    const int  bA   = b0 + r;

    if (tid < 96) stat_lds[tid] = 0.0f;
    __syncthreads();

    // ---- B fragments (K=32, N=16). Lane: N=n, K = lo ? {0..7,16..23} : {8..15,24..31}.
    // frag[i] = B[kb+i][N] for i<8, B[16+kb+i-8][N] for i>=8.
    //   B_r [k][N] = k<16 ? Wih[N][k]      : Whh[N][k-16]
    //   B_z [k][N] = k<16 ? Wih[16+N][k]   : Whh[16+N][k-16]
    //   B_xn[k][N] = k<16 ? Wih[32+N][k]   : 0
    //   B_hn[k][N] = k<16 ? 0              : Whh[32+N][k-16]
    v16h Brf, Bzf, Bxnf, Bhnf;
    {
        const int kb = lo ? 0 : 8;
        #pragma unroll
        for (int i = 0; i < 16; ++i) {
            Brf[i]  = (_Float16)(i < 8 ? Wih[n * 16 + kb + i]        : Whh[n * 16 + kb + i - 8]);
            Bzf[i]  = (_Float16)(i < 8 ? Wih[(16 + n) * 16 + kb + i] : Whh[(16 + n) * 16 + kb + i - 8]);
            Bxnf[i] = i < 8 ? (_Float16)Wih[(32 + n) * 16 + kb + i] : (_Float16)0.0f;
            Bhnf[i] = i < 8 ? (_Float16)0.0f : (_Float16)Whh[(32 + n) * 16 + kb + i - 8];
        }
    }

    // Per-lane gate biases (column n of each 16-wide gate tile).
    const float bias_r  = bih[n]      + bhh[n];
    const float bias_z  = bih[16 + n] + bhh[16 + n];
    const float bias_xn = bih[32 + n];
    const float bias_hn = bhh[32 + n];

    // h = 0 in registers (C-layout: 8 batch rows per lane) and in the LDS mirror.
    float h[8], hsum[8];
    #pragma unroll
    for (int m = 0; m < 8; ++m) { h[m] = 0.0f; hsum[m] = 0.0f; }
    #pragma unroll
    for (int m = 0; m < 8; ++m) {
        int M = lo ? m : (m + 8);
        h_lds[wave][M][n] = (_Float16)0.0f;
    }

    const v8f c0 = {};

    for (int t = 0; t < T; ++t) {
        // ---- A fragment: rows = batch, K0..15 = x (item||cat emb), K16..31 = h
        v16h a;
        {
            const int idx  = hist_item[(size_t)bA * T + t];
            const int cidx = hist_cat [(size_t)bA * T + t];
            const float* itrow = item_table + (size_t)idx * 12;   // 48B rows: 16B aligned
            const float4* p0 = (const float4*)(lo ? itrow       : (itrow + 8));
            const float4* p1 = (const float4*)(lo ? (itrow + 4) : (cat_table + (size_t)cidx * 4));
            float4 x0 = *p0;
            float4 x1 = *p1;
            a[0] = (_Float16)x0.x; a[1] = (_Float16)x0.y;
            a[2] = (_Float16)x0.z; a[3] = (_Float16)x0.w;
            a[4] = (_Float16)x1.x; a[5] = (_Float16)x1.y;
            a[6] = (_Float16)x1.z; a[7] = (_Float16)x1.w;
            const v8h hv = *(const v8h*)&h_lds[wave][r][lo ? 0 : 8];
            #pragma unroll
            for (int i = 0; i < 8; ++i) a[8 + i] = hv[i];
        }

        // ---- gates: D = A x B (C=0), biases added in the elementwise stage
        v8f dr  = __builtin_amdgcn_wmma_f32_16x16x32_f16(false, a, false, Brf,  (short)0, c0, false, false);
        v8f dz  = __builtin_amdgcn_wmma_f32_16x16x32_f16(false, a, false, Bzf,  (short)0, c0, false, false);
        v8f dxn = __builtin_amdgcn_wmma_f32_16x16x32_f16(false, a, false, Bxnf, (short)0, c0, false, false);
        v8f dhn = __builtin_amdgcn_wmma_f32_16x16x32_f16(false, a, false, Bhnf, (short)0, c0, false, false);

        // ---- GRU cell update (lane-local in the C/D layout) + LDS writeback
        #pragma unroll
        for (int m = 0; m < 8; ++m) {
            float rg = sigmf(dr[m] + bias_r);
            float zg = sigmf(dz[m] + bias_z);
            float ng = tanhfast(dxn[m] + bias_xn + rg * (dhn[m] + bias_hn));
            float hv = ng + zg * (h[m] - ng);          // (1-z)*n + z*h
            h[m] = hv;
            hsum[m] += hv;
            int M = lo ? m : (m + 8);
            h_lds[wave][M][n] = (_Float16)hv;
        }
    }

    // ---- stage target embedding tile (same gather pattern as x)
    {
        const int idx  = tgt_item[bA];
        const int cidx = tgt_cat [bA];
        const float* itrow = item_table + (size_t)idx * 12;
        const float4* p0 = (const float4*)(lo ? itrow       : (itrow + 8));
        const float4* p1 = (const float4*)(lo ? (itrow + 4) : (cat_table + (size_t)cidx * 4));
        float4 x0 = *p0;
        float4 x1 = *p1;
        float* dst = &tgt_lds[wave][r][lo ? 0 : 8];
        dst[0] = x0.x; dst[1] = x0.y; dst[2] = x0.z; dst[3] = x0.w;
        dst[4] = x1.x; dst[5] = x1.y; dst[6] = x1.z; dst[7] = x1.w;
    }

    // ---- degenerate attention: out_v[b][n] = sum_i tgt[b][i]*Weff[i] + beff
    // with Weff = Wout @ Wv, beff = Wout @ bv + bout  (softmax over S=1 is 1.0)
    float weff[16];
    #pragma unroll
    for (int i = 0; i < 16; ++i) weff[i] = 0.0f;
    float beff = bout[n];
    for (int d = 0; d < 16; ++d) {
        float wo = Wout[n * 16 + d];
        beff += wo * binp[32 + d];
        const float* wvrow = Win + (32 + d) * 16;   // Wv rows of packed in_proj
        #pragma unroll
        for (int i = 0; i < 16; ++i) weff[i] += wo * wvrow[i];
    }

    float sa = 0.f, sa2 = 0.f, st = 0.f, st2 = 0.f, su = 0.f, su2 = 0.f;
    #pragma unroll
    for (int m = 0; m < 8; ++m) {
        int M = lo ? m : (m + 8);
        int b = b0 + M;
        float ov = beff;
        #pragma unroll
        for (int i = 0; i < 16; ++i) ov += tgt_lds[wave][M][i] * weff[i];
        float avg = ov * hsum[m];                       // (attn * hist_enc).sum(t)
        float tg  = tgt_lds[wave][M][n];
        float uv  = user_table[(size_t)user_id[b] * 16 + n];
        all_enc[(size_t)b * 48 + n]      = avg;
        all_enc[(size_t)b * 48 + 16 + n] = tg;
        all_enc[(size_t)b * 48 + 32 + n] = uv;
        sa += avg; sa2 += avg * avg;
        st += tg;  st2 += tg * tg;
        su += uv;  su2 += uv * uv;
    }
    atomicAdd(&stat_lds[n],            sa);
    atomicAdd(&stat_lds[48 + n],       sa2);
    atomicAdd(&stat_lds[16 + n],       st);
    atomicAdd(&stat_lds[48 + 16 + n],  st2);
    atomicAdd(&stat_lds[32 + n],       su);
    atomicAdd(&stat_lds[48 + 32 + n],  su2);
    __syncthreads();
    if (tid < 96) atomicAdd(&g_stats1[tid], stat_lds[tid]);
}

// -----------------------------------------------------------------------------
// Kernel 2: BN1 (batch stats) -> FC1(48->16) -> ReLU; accumulate BN2 stats.
// -----------------------------------------------------------------------------
__global__ void __launch_bounds__(256)
k_fc1(const float* __restrict__ all_enc, const float* __restrict__ g_stats1,
      const float* __restrict__ bn1_g, const float* __restrict__ bn1_b,
      const float* __restrict__ fc1_W, const float* __restrict__ fc1_b,
      float* __restrict__ h1, float* __restrict__ g_stats2,
      float invB, int B)
{
    __shared__ float s2[32];
    int tid = threadIdx.x;
    if (tid < 32) s2[tid] = 0.0f;
    __syncthreads();
    int b = blockIdx.x * 256 + tid;
    if (b < B) {
        float acc[16];
        #pragma unroll
        for (int j = 0; j < 16; ++j) acc[j] = fc1_b[j];
        for (int c = 0; c < 48; ++c) {
            float mu  = g_stats1[c] * invB;
            float var = g_stats1[48 + c] * invB - mu * mu;
            float x = (all_enc[(size_t)b * 48 + c] - mu) * rsqrtf(var + EPSBN)
                      * bn1_g[c] + bn1_b[c];
            #pragma unroll
            for (int j = 0; j < 16; ++j) acc[j] += x * fc1_W[j * 48 + c];
        }
        #pragma unroll
        for (int j = 0; j < 16; ++j) {
            float v = fmaxf(acc[j], 0.0f);
            h1[(size_t)b * 16 + j] = v;
            atomicAdd(&s2[j], v);
            atomicAdd(&s2[16 + j], v * v);
        }
    }
    __syncthreads();
    if (tid < 32) atomicAdd(&g_stats2[tid], s2[tid]);
}

// -----------------------------------------------------------------------------
// Kernel 3: BN2 (batch stats) -> FC2(16->2) -> softmax.
// -----------------------------------------------------------------------------
__global__ void __launch_bounds__(256)
k_fc2(const float* __restrict__ h1, const float* __restrict__ g_stats2,
      const float* __restrict__ bn2_g, const float* __restrict__ bn2_b,
      const float* __restrict__ fc2_W, const float* __restrict__ fc2_b,
      float* __restrict__ out, float invB, int B)
{
    int b = blockIdx.x * 256 + threadIdx.x;
    if (b >= B) return;
    float l0 = fc2_b[0], l1 = fc2_b[1];
    for (int j = 0; j < 16; ++j) {
        float mu  = g_stats2[j] * invB;
        float var = g_stats2[16 + j] * invB - mu * mu;
        float x = (h1[(size_t)b * 16 + j] - mu) * rsqrtf(var + EPSBN)
                  * bn2_g[j] + bn2_b[j];
        l0 += x * fc2_W[j];
        l1 += x * fc2_W[16 + j];
    }
    float mx = fmaxf(l0, l1);
    float e0 = __expf(l0 - mx), e1 = __expf(l1 - mx);
    float rs = __builtin_amdgcn_rcpf(e0 + e1);
    out[(size_t)b * 2 + 0] = e0 * rs;
    out[(size_t)b * 2 + 1] = e1 * rs;
}

__global__ void k_zero(float* p, int nn) {
    int i = blockIdx.x * 256 + threadIdx.x;
    if (i < nn) p[i] = 0.0f;
}

// -----------------------------------------------------------------------------
extern "C" void kernel_launch(void* const* d_in, const int* in_sizes, int n_in,
                              void* d_out, int out_size, void* d_ws, size_t ws_size,
                              hipStream_t stream)
{
    (void)n_in; (void)out_size; (void)ws_size;
    const int*   user_id    = (const int*)  d_in[0];
    const int*   hist_item  = (const int*)  d_in[1];
    const int*   hist_cat   = (const int*)  d_in[2];
    const int*   tgt_item   = (const int*)  d_in[3];
    const int*   tgt_cat    = (const int*)  d_in[4];
    const float* user_table = (const float*)d_in[5];
    const float* item_table = (const float*)d_in[6];
    const float* cat_table  = (const float*)d_in[7];
    const float* Wih        = (const float*)d_in[8];
    const float* Whh        = (const float*)d_in[9];
    const float* bih        = (const float*)d_in[10];
    const float* bhh        = (const float*)d_in[11];
    const float* Win        = (const float*)d_in[12];
    const float* binp       = (const float*)d_in[13];
    const float* Wout       = (const float*)d_in[14];
    const float* bout       = (const float*)d_in[15];
    const float* bn1_g      = (const float*)d_in[16];
    const float* bn1_b      = (const float*)d_in[17];
    const float* fc1_W      = (const float*)d_in[18];
    const float* fc1_b      = (const float*)d_in[19];
    const float* bn2_g      = (const float*)d_in[20];
    const float* bn2_b      = (const float*)d_in[21];
    const float* fc2_W      = (const float*)d_in[22];
    const float* fc2_b      = (const float*)d_in[23];

    const int B = in_sizes[0];
    const int T = in_sizes[1] / B;

    float* all_enc  = (float*)d_ws;                  // B*48
    float* h1       = all_enc + (size_t)B * 48;      // B*16
    float* g_stats1 = h1 + (size_t)B * 16;           // 96 floats
    float* g_stats2 = g_stats1 + 96;                 // 32 floats
    const float invB = 1.0f / (float)B;

    k_zero<<<1, 256, 0, stream>>>(g_stats1, 128);
    k_gru_attn<<<B / (16 * WPB), GRUBLK, 0, stream>>>(
        user_id, hist_item, hist_cat, tgt_item, tgt_cat,
        user_table, item_table, cat_table,
        Wih, Whh, bih, bhh, Win, binp, Wout, bout,
        all_enc, g_stats1, B, T);
    k_fc1<<<(B + 255) / 256, 256, 0, stream>>>(
        all_enc, g_stats1, bn1_g, bn1_b, fc1_W, fc1_b, h1, g_stats2, invB, B);
    k_fc2<<<(B + 255) / 256, 256, 0, stream>>>(
        h1, g_stats2, bn2_g, bn2_b, fc2_W, fc2_b, (float*)d_out, invB, B);
}
